// QConv2d_24781961298594
// MI455X (gfx1250) — 8-bit matrix operand order repaired and checked by reading, compiled, not run on hardware
//
#include <hip/hip_runtime.h>
#include <stdint.h>

typedef int v8i __attribute__((ext_vector_type(8)));

#define NB    32
#define C_IN  64
#define H_IN  128
#define W_IN  128
#define O_CH  64
#define H_OUT 64
#define W_OUT 64

__device__ __forceinline__ void async_copy_b128(const void* gptr, void* lptr) {
    uint32_t lds_off = (uint32_t)(uintptr_t)lptr;
    asm volatile("global_load_async_to_lds_b128 %0, %1, off"
                 :: "v"(lds_off), "v"(gptr)
                 : "memory");
}
__device__ __forceinline__ void wait_asynccnt0() {
    asm volatile("s_wait_asynccnt 0x0" ::: "memory");
}

__global__ void __launch_bounds__(256) binarize_x_kernel(const float* __restrict__ x,
                                                         int8_t* __restrict__ xb) {
    __shared__ int lds[W_IN * C_IN / 4];
    int8_t* lds_b = (int8_t*)lds;
    const int n   = blockIdx.x >> 7;
    const int h   = blockIdx.x & 127;
    const int tid = threadIdx.x;
    const float* src = x + ((size_t)n * C_IN * H_IN + (size_t)h) * W_IN;
#pragma unroll
    for (int it = 0; it < 32; ++it) {
        int li = it * 256 + tid;
        int c  = li >> 7;
        int w  = li & 127;
        float v = __builtin_nontemporal_load(&src[(size_t)c * H_IN * W_IN + w]);
        lds_b[w * C_IN + c] = (v >= 0.0f) ? (int8_t)1 : (int8_t)-1;
    }
    __syncthreads();
    int4* dst = (int4*)(xb + ((size_t)n * H_IN + h) * (size_t)(W_IN * C_IN));
    const int4* s4 = (const int4*)lds;
    dst[tid]       = s4[tid];
    dst[256 + tid] = s4[256 + tid];
}

__global__ void __launch_bounds__(256) binarize_w_kernel(const float* __restrict__ w,
                                                         int8_t* __restrict__ wb) {
    int idx = blockIdx.x * 256 + threadIdx.x;
    if (idx >= O_CH * C_IN * 9) return;
    int o   = idx / (C_IN * 9);
    int rem = idx % (C_IN * 9);
    int c   = rem / 9;
    int tap = rem % 9;
    float v = w[idx];
    wb[(tap * O_CH + o) * C_IN + c] = (v >= 0.0f) ? (int8_t)1 : (int8_t)-1;
}

__global__ void __launch_bounds__(256) bconv_wmma_kernel(const int8_t* __restrict__ xb,
                                                         const int8_t* __restrict__ wb,
                                                         float* __restrict__ out) {
    const int n    = blockIdx.x >> 6;
    const int ho   = blockIdx.x & 63;
    const int tid  = threadIdx.x;
    const int lane = tid & 31;
    const int wave = tid >> 5;

    __shared__ int lds_w[9 * O_CH * 16];
    __shared__ int lds_x[3 * 132 * 16];

    {
        int4* d = (int4*)lds_w;
        const int4* s = (const int4*)wb;
#pragma unroll
        for (int it = 0; it < 9; ++it) d[it * 256 + tid] = s[it * 256 + tid];
    }
#pragma unroll
    for (int r = 0; r < 3; ++r) {
        int hi = 2 * ho - 1 + r;
        int8_t* drow = (int8_t*)(lds_x + r * 2112 + 16);
        if (hi >= 0) {
            const int8_t* srow = xb + ((size_t)n * H_IN + hi) * (size_t)(W_IN * C_IN);
            async_copy_b128(srow + (size_t)tid * 16,        drow + tid * 16);
            async_copy_b128(srow + 4096 + (size_t)tid * 16, drow + 4096 + tid * 16);
        } else {
            int4 z; z.x = z.y = z.z = z.w = 0;
            int4* d = (int4*)drow;
            d[tid]       = z;
            d[256 + tid] = z;
        }
    }
    if (tid < 48) {
        int r = tid >> 4;
        lds_x[r * 2112 + (tid & 15)] = 0;
    }
    wait_asynccnt0();
    __syncthreads();

    const int m      = wave >> 1;
    const int obase  = m * 16;
    const int wbase0 = ((wave & 1) << 1) * 16;
    const int wbase1 = wbase0 + 16;

    const int aM    = obase + (lane & 15);
    const int koffA = (lane & 16) ? 8 : 0;
    const int col   = lane & 15;
    const int koffB = (lane & 16) ? 16 : 0;

    v8i acc0 = {0, 0, 0, 0, 0, 0, 0, 0};
    v8i acc1 = {0, 0, 0, 0, 0, 0, 0, 0};

#pragma unroll
    for (int kh = 0; kh < 3; ++kh) {
#pragma unroll
        for (int kw = 0; kw < 3; ++kw) {
            const int tap = kh * 3 + kw;
            v8i a;
            const int abase = tap * 1024 + aM * 16;
#pragma unroll
            for (int v = 0; v < 8; ++v) {
                int K = ((v >> 1) << 4) + ((v & 1) << 2) + koffA;
                a[v] = lds_w[abase + (K >> 2)];
            }
            const int wi0    = 2 * (wbase0 + col) - 1 + kw;
            const int b0base = kh * 2112 + (wi0 + 1) * 16;
            const int b1base = b0base + 32 * 16;
            v8i b0, b1;
#pragma unroll
            for (int v = 0; v < 8; ++v) {
                int K = ((v >> 1) << 4) + ((v & 1) << 2) + koffA;
                b0[v] = lds_x[b0base + (K >> 2)];
                b1[v] = lds_x[b1base + (K >> 2)];
            }
            acc0 = __builtin_amdgcn_wmma_i32_16x16x64_iu8(true, a, true, b0, acc0, false, false);
            acc1 = __builtin_amdgcn_wmma_i32_16x16x64_iu8(true, a, true, b1, acc1, false, false);
        }
    }

    const int oAdd = (lane & 16) ? 8 : 0;
    float* outp = out + ((size_t)n * O_CH * H_OUT + (size_t)ho) * W_OUT;
#pragma unroll
    for (int r = 0; r < 8; ++r) {
        int o = obase + r + oAdd;
        __builtin_nontemporal_store((float)acc0[r],
            &outp[(size_t)o * H_OUT * W_OUT + wbase0 + col]);
        __builtin_nontemporal_store((float)acc1[r],
            &outp[(size_t)o * H_OUT * W_OUT + wbase1 + col]);
    }
}

extern "C" void kernel_launch(void* const* d_in, const int* in_sizes, int n_in,
                              void* d_out, int out_size, void* d_ws, size_t ws_size,
                              hipStream_t stream) {
    const float* x = (const float*)d_in[0];
    const float* w = (const float*)d_in[1];
    float* out = (float*)d_out;

    int8_t* xb = (int8_t*)d_ws;
    int8_t* wb = xb + (size_t)NB * H_IN * W_IN * C_IN;

    binarize_x_kernel<<<NB * H_IN, 256, 0, stream>>>(x, xb);
    binarize_w_kernel<<<(O_CH * C_IN * 9 + 255) / 256, 256, 0, stream>>>(w, wb);
    bconv_wmma_kernel<<<NB * H_OUT, 256, 0, stream>>>(xb, wb, out);
}
